// GGNN_16449724744291
// MI455X (gfx1250) — matrix operand order repaired and checked by reading, not compiled, not run on hardware
//
#include <hip/hip_runtime.h>
#include <hip/hip_bf16.h>

typedef __attribute__((ext_vector_type(16))) _Float16 v16h;
typedef __attribute__((ext_vector_type(8)))  float    v8f;

#define WAVES 8
#define NB 16
#define ROWS_O 160
#define ROWS_D 176

// Packed-weight tile offsets in workspace (units of 256 dwords = one 32x16 f16 B tile)
enum {
  T_WG_R  = 0,    // 2x4
  T_WG_O  = 8,    // 3x4
  T_WG_D  = 20,   // 3x4
  T_WZR_R = 32,   // 4x8
  T_WZR_O = 64,   // 6x8
  T_WZR_D = 112,  // 4x8
  T_WM_R  = 144,  // 4x4
  T_WM_O  = 160,  // 6x4
  T_WM_D  = 184,  // 4x4
  T_W1_R  = 200,  // 2x1
  T_W1_O  = 202,
  T_W1_D  = 204,
  T_TOTAL = 206
};

union Frag { v16h v; uint4 q[2]; };

// B fragment (32x16 f16): lane holds N=lane&15, half = lane>>4,
// dword j packs K = (j<4?0:16)+2(j&3)+8*half (lo), +1 (hi). Pre-packed so this is 2x b128.
__device__ __forceinline__ v16h load_b_tile(const unsigned* w, int tile, int lane) {
  Frag f;
  const uint4* p = (const uint4*)(w + tile * 256 + lane * 8);
  f.q[0] = p[0];
  f.q[1] = p[1];
  return f.v;
}

// A fragment (16x32 f16) from LDS rows [row][K], row stride `stride` halfs.
// Lane: M = m0 + (lane&15), half = lane>>4; VGPR j: K = (j<4?0:16)+2(j&3)+8*half.
__device__ __forceinline__ v16h load_a_tile(const _Float16* a, int m0, int stride,
                                            int kt, int lane) {
  int m = m0 + (lane & 15);
  int half = lane >> 4;
  const _Float16* r = a + m * stride + kt * 32 + 8 * half;
  Frag f;
  f.q[0] = *(const uint4*)(r);
  f.q[1] = *(const uint4*)(r + 16);
  return f.v;
}

// Fast activations: gfx1250 has native v_tanh_f32 / v_exp_f32 / v_rcp_f32.
__device__ __forceinline__ float fast_tanh(float x) {
#if __has_builtin(__builtin_amdgcn_tanhf)
  return __builtin_amdgcn_tanhf(x);
#elif __has_builtin(__builtin_amdgcn_tanh_f32)
  return __builtin_amdgcn_tanh_f32(x);
#else
  return tanhf(x);
#endif
}
__device__ __forceinline__ float fast_rcp(float x) {
#if __has_builtin(__builtin_amdgcn_rcpf)
  return __builtin_amdgcn_rcpf(x);
#else
  return 1.f / x;
#endif
}
__device__ __forceinline__ float fast_sigmoid(float x) {
  return fast_rcp(1.f + __expf(-x));
}
__device__ __forceinline__ float sigmoid_(float x) { return 1.f / (1.f + expf(-x)); }

// Fully-unrolled tile GEMM, two A sources (h-block then n-block): load all
// fragments first (one load clause), then the WMMA accumulation chain.
template <int K1, int K2>
__device__ __forceinline__ v8f gemm2(const _Float16* a1, int s1,
                                     const _Float16* a2, int s2, int m0,
                                     const unsigned* wt, int wb, int NT, int nt,
                                     int lane) {
  v16h A[K1 + K2];
  v16h B[K1 + K2];
#pragma unroll
  for (int k = 0; k < K1; k++) {
    A[k] = load_a_tile(a1, m0, s1, k, lane);
    B[k] = load_b_tile(wt, wb + k * NT + nt, lane);
  }
#pragma unroll
  for (int k = 0; k < K2; k++) {
    A[K1 + k] = load_a_tile(a2, m0, s2, k, lane);
    B[K1 + k] = load_b_tile(wt, wb + (K1 + k) * NT + nt, lane);
  }
  v8f c = {};
#pragma unroll
  for (int k = 0; k < K1 + K2; k++)
    c = __builtin_amdgcn_wmma_f32_16x16x32_f16(false, A[k], false, B[k], (short)0, c,
                                               false, false);
  return c;
}

template <int KT>
__device__ __forceinline__ v8f gemm1(const _Float16* a, int s, int m0,
                                     const unsigned* wt, int wb, int NT, int nt,
                                     int lane) {
  v16h A[KT];
  v16h B[KT];
#pragma unroll
  for (int k = 0; k < KT; k++) {
    A[k] = load_a_tile(a, m0, s, k, lane);
    B[k] = load_b_tile(wt, wb + k * NT + nt, lane);
  }
  v8f c = {};
#pragma unroll
  for (int k = 0; k < KT; k++)
    c = __builtin_amdgcn_wmma_f32_16x16x32_f16(false, A[k], false, B[k], (short)0, c,
                                               false, false);
  return c;
}

// ---------------------------------------------------------------------------
// Weight packing: f32 (K x Ncols[, dual source side-by-side in N]) -> f16 B tiles
// ---------------------------------------------------------------------------
__global__ void ggnn_pack_b(const float* __restrict__ s0, const float* __restrict__ s1,
                            int K, int Ncols, int NT, unsigned* __restrict__ dst,
                            int total) {
  int id = blockIdx.x * 256 + threadIdx.x;
  if (id >= total) return;
  int j    = id & 7;
  int lane = (id >> 3) & 31;
  int t    = id >> 8;
  int nt   = t % NT;
  int kt   = t / NT;
  int n    = nt * 16 + (lane & 15);
  int half = lane >> 4;
  int k0   = kt * 32 + 8 * half + 16 * (j >> 2) + 2 * (j & 3);
  const float* src = s0;
  int col = n;
  if (s1 != nullptr && n >= Ncols) { src = s1; col = n - Ncols; }
  float lo = (k0     < K) ? src[(k0    ) * Ncols + col] : 0.f;
  float hi = (k0 + 1 < K) ? src[(k0 + 1) * Ncols + col] : 0.f;
  union { _Float16 h[2]; unsigned u; } cv;
  cv.h[0] = (_Float16)lo;
  cv.h[1] = (_Float16)hi;
  dst[t * 256 + lane * 8 + j] = cv.u;
}

// ---------------------------------------------------------------------------
// Temporal bidirectional mini-GRU scan + softmax + gather (3 channels)
// ---------------------------------------------------------------------------
__global__ void ggnn_temporal(const float* __restrict__ temporal,
                              const float* __restrict__ gk,
                              const float* __restrict__ grk,
                              const float* __restrict__ gb,
                              const float* __restrict__ wo,
                              const int* __restrict__ idx,
                              float* __restrict__ tp) {
  int ch = threadIdx.x;
  if (ch >= 3) return;
  float f[75], g[75];
  const float* t = temporal + ch * 75;
  {
    float k0 = gk[ch*6+0], k1 = gk[ch*6+1], k2 = gk[ch*6+2];
    float r0 = grk[ch*6+0], r1 = grk[ch*6+1], r2 = grk[ch*6+2];
    const float* b = gb + ch * 12;          // scan 0: b[p*3+i]
    float h = 0.f;
    for (int s = 0; s < 75; s++) {
      float x  = t[s];
      float z  = sigmoid_(x*k0 + b[0] + h*r0 + b[3]);
      float r  = sigmoid_(x*k1 + b[1] + h*r1 + b[4]);
      float hh = tanhf(x*k2 + b[2] + r*(h*r2 + b[5]));
      h = z*h + (1.f - z)*hh;
      f[s] = h;
    }
  }
  {
    float k0 = gk[ch*6+3], k1 = gk[ch*6+4], k2 = gk[ch*6+5];
    float r0 = grk[ch*6+3], r1 = grk[ch*6+4], r2 = grk[ch*6+5];
    const float* b = gb + ch * 12 + 6;      // scan 1
    float h = 0.f;
    for (int s = 0; s < 75; s++) {
      float x  = f[74 - s];
      float z  = sigmoid_(x*k0 + b[0] + h*r0 + b[3]);
      float r  = sigmoid_(x*k1 + b[1] + h*r1 + b[4]);
      float hh = tanhf(x*k2 + b[2] + r*(h*r2 + b[5]));
      h = z*h + (1.f - z)*hh;
      g[s] = h;
    }
  }
  float w = wo[ch];
  float mx = -3.0e38f;
  for (int s = 0; s < 75; s++) { g[s] *= w; mx = fmaxf(mx, g[s]); }
  float sum = 0.f;
  for (int s = 0; s < 75; s++) { g[s] = expf(g[s] - mx); sum += g[s]; }
  float inv = 1.f / sum;
  for (int j = 0; j < 60; j++) tp[ch * 60 + j] = g[idx[j]] * inv;
}

// ---------------------------------------------------------------------------
// Main GGNN kernel: 1 workgroup = 16 batch elements, 256 threads (8 wave32)
// ---------------------------------------------------------------------------
struct MainParams {
  const float *X, *A;
  const float *bgR, *bgO, *bgD;
  const float *bmR, *bmO, *bmD;
  const float *b1R, *b1O, *b1D;
  const float *w2R, *w2O, *w2D;
  const unsigned* wt;   // packed weight tiles
  float *sR, *sO, *sD;
};

#define SMEM_BYTES 238592

__global__ __launch_bounds__(256) void ggnn_main(MainParams p) {
  extern __shared__ char smem[];
  _Float16* hR  = (_Float16*)(smem + 0);        // 16x64
  _Float16* hO  = (_Float16*)(smem + 2048);     // 160x64
  _Float16* hD  = (_Float16*)(smem + 22528);    // 176x64
  _Float16* nR  = (_Float16*)(smem + 45056);    // 16x64   (init: XR 16x64)
  _Float16* nO  = (_Float16*)(smem + 47104);    // 160x128 (init: XO 160x96)
  _Float16* nD  = (_Float16*)(smem + 88064);    // 176x64
  _Float16* hrR = (_Float16*)(smem + 110592);   // 16x64
  _Float16* hrO = (_Float16*)(smem + 112640);   // 160x64  (init: XD 176x96 spans hrO+hrD)
  _Float16* hrD = (_Float16*)(smem + 133120);   // 176x64
  _Float16* zR  = (_Float16*)(smem + 155648);   // 16x64
  _Float16* zO  = (_Float16*)(smem + 157696);   // 160x64
  _Float16* zD  = (_Float16*)(smem + 178176);   // 176x64
  float*    aRO = (float*)(smem + 200704);      // 16x10
  float*    aOR = (float*)(smem + 201344);      // 16x10
  float*    aOD = (float*)(smem + 201984);      // 16x10x11
  float*    aDO = (float*)(smem + 209024);      // 16x11x10
  float*    rd  = (float*)(smem + 216064);      // 352x16

  const int tid  = threadIdx.x;
  const int lane = tid & 31;
  const int wid  = tid >> 5;
  const int b0   = blockIdx.x * NB;
  _Float16* XD = hrO;

  // ---- stage X (f16, zero-padded) and A sub-blocks ----
  for (int e = tid; e < 16 * 64; e += 256) {
    int r = e >> 6, c = e & 63;
    float v = (c < 40) ? p.X[((b0 + r) * 22 + 0) * 40 + c] : 0.f;
    nR[e] = (_Float16)v;
  }
  for (int e = tid; e < ROWS_O * 96; e += 256) {
    int r = e / 96, c = e - r * 96;
    int bl = r / 10, o = r - bl * 10;
    const float* xb = p.X + (b0 + bl) * 22 * 40;
    float v = 0.f;
    if (c < 40)      v = xb[c];
    else if (c < 70) v = xb[(1 + o) * 40 + 10 + (c - 40)];
    nO[r * 96 + c] = (_Float16)v;
  }
  for (int e = tid; e < ROWS_D * 96; e += 256) {
    int r = e / 96, c = e - r * 96;
    int bl = r / 11, d = r - bl * 11;
    const float* xb = p.X + (b0 + bl) * 22 * 40;
    float v = 0.f;
    if (c < 40)      v = xb[c];
    else if (c < 70) v = xb[(11 + d) * 40 + 10 + (c - 40)];
    XD[r * 96 + c] = (_Float16)v;
  }
  for (int e = tid; e < 160; e += 256) {
    int bl = e / 10, o = e - bl * 10;
    aRO[e] = p.A[((b0 + bl) * 22 + 0) * 22 + 1 + o];
    aOR[e] = p.A[((b0 + bl) * 22 + 1 + o) * 22 + 0];
  }
  for (int e = tid; e < 1760; e += 256) {
    int bl = e / 110, rem = e - bl * 110;
    { int o = rem / 11, d = rem - o * 11;
      aOD[e] = p.A[((b0 + bl) * 22 + 1 + o) * 22 + 11 + d]; }
    { int d = rem / 10, o = rem - d * 10;
      aDO[(bl * 11 + d) * 10 + o] = p.A[((b0 + bl) * 22 + 11 + d) * 22 + 1 + o]; }
  }
  __syncthreads();

  // ---- init: h = tanh(X @ Wg + bg) ----
  for (int job = wid; job < 88; job += WAVES) {
    _Float16* out; const float* bias;
    int m0, nt;
    v8f c;
    if (job < 4) {
      nt = job; m0 = 0;
      c = gemm1<2>(nR, 64, m0, p.wt, T_WG_R, 4, nt, lane);
      bias = p.bgR; out = hR;
    } else if (job < 44) {
      int i = job - 4; int mt = i >> 2; nt = i & 3; m0 = mt * 16;
      c = gemm1<3>(nO, 96, m0, p.wt, T_WG_O, 4, nt, lane);
      bias = p.bgO; out = hO;
    } else {
      int i = job - 44; int mt = i >> 2; nt = i & 3; m0 = mt * 16;
      c = gemm1<3>(XD, 96, m0, p.wt, T_WG_D, 4, nt, lane);
      bias = p.bgD; out = hD;
    }
    int half = lane >> 4, nG = nt * 16 + (lane & 15);
    float bv = bias[nG];
#pragma unroll
    for (int r = 0; r < 8; r++) {
      int row = m0 + r + 8 * half;
      out[row * 64 + nG] = (_Float16)fast_tanh(c[r] + bv);
    }
  }
  __syncthreads();

  // ---- 3 GRU layers ----
  for (int layer = 0; layer < 3; layer++) {
    // message passing (VALU, ~2% of FLOPs)
    for (int e = tid; e < 32768; e += 256) {
      if (e < 1024) {                                  // n_R = A_RO @ h_O
        int bl = e >> 6, c = e & 63;
        float acc = 0.f;
        for (int o = 0; o < 10; o++) acc += aRO[bl * 10 + o] * (float)hO[(bl * 10 + o) * 64 + c];
        nR[e] = (_Float16)acc;
      } else if (e < 11264) {                          // n_O[:, :64] = A_OR @ h_R
        int t = e - 1024; int row = t >> 6, c = t & 63; int bl = row / 10;
        nO[row * 128 + c] = (_Float16)(aOR[row] * (float)hR[bl * 64 + c]);
      } else if (e < 21504) {                          // n_O[:, 64:] = A_OD @ h_D
        int t = e - 11264; int row = t >> 6, c = t & 63; int bl = row / 10;
        float acc = 0.f;
        for (int d = 0; d < 11; d++) acc += aOD[row * 11 + d] * (float)hD[(bl * 11 + d) * 64 + c];
        nO[row * 128 + 64 + c] = (_Float16)acc;
      } else {                                         // n_D = A_DO @ h_O
        int t = e - 21504; int row = t >> 6, c = t & 63; int bl = row / 11;
        float acc = 0.f;
        for (int o = 0; o < 10; o++) acc += aDO[row * 10 + o] * (float)hO[(bl * 10 + o) * 64 + c];
        nD[row * 64 + c] = (_Float16)acc;
      }
    }
    __syncthreads();

    // z/r: sigmoid([h|n] @ [Wz|Wr]); also stage hr = h * r
    for (int job = wid; job < 176; job += WAVES) {
      const _Float16* hb; _Float16 *zb, *hrb;
      int m0, nt;
      v8f c;
      if (job < 8) {
        nt = job; m0 = 0; hb = hR; zb = zR; hrb = hrR;
        c = gemm2<2, 2>(hR, 64, nR, 64, m0, p.wt, T_WZR_R, 8, nt, lane);
      } else if (job < 88) {
        int i = job - 8; int mt = i >> 3; nt = i & 7; m0 = mt * 16;
        hb = hO; zb = zO; hrb = hrO;
        c = gemm2<2, 4>(hO, 64, nO, 128, m0, p.wt, T_WZR_O, 8, nt, lane);
      } else {
        int i = job - 88; int mt = i >> 3; nt = i & 7; m0 = mt * 16;
        hb = hD; zb = zD; hrb = hrD;
        c = gemm2<2, 2>(hD, 64, nD, 64, m0, p.wt, T_WZR_D, 8, nt, lane);
      }
      int half = lane >> 4, nG = nt * 16 + (lane & 15);
#pragma unroll
      for (int r = 0; r < 8; r++) {
        int row = m0 + r + 8 * half;
        float s = fast_sigmoid(c[r]);
        if (nG < 64) zb[row * 64 + nG] = (_Float16)s;
        else {
          int cc = nG - 64;
          hrb[row * 64 + cc] = (_Float16)(s * (float)hb[row * 64 + cc]);
        }
      }
    }
    __syncthreads();

    // m: hm = tanh([h*r|n] @ Wm + bm); h = (1-z)h + z*hm
    for (int job = wid; job < 88; job += WAVES) {
      _Float16* hb; const _Float16* zb; const float* bm;
      int m0, nt;
      v8f c;
      if (job < 4) {
        nt = job; m0 = 0; hb = hR; zb = zR; bm = p.bmR;
        c = gemm2<2, 2>(hrR, 64, nR, 64, m0, p.wt, T_WM_R, 4, nt, lane);
      } else if (job < 44) {
        int i = job - 4; int mt = i >> 2; nt = i & 3; m0 = mt * 16;
        hb = hO; zb = zO; bm = p.bmO;
        c = gemm2<2, 4>(hrO, 64, nO, 128, m0, p.wt, T_WM_O, 4, nt, lane);
      } else {
        int i = job - 44; int mt = i >> 2; nt = i & 3; m0 = mt * 16;
        hb = hD; zb = zD; bm = p.bmD;
        c = gemm2<2, 2>(hrD, 64, nD, 64, m0, p.wt, T_WM_D, 4, nt, lane);
      }
      int half = lane >> 4, nG = nt * 16 + (lane & 15);
      float bv = bm[nG];
#pragma unroll
      for (int r = 0; r < 8; r++) {
        int row = m0 + r + 8 * half;
        float hm = fast_tanh(c[r] + bv);
        float z  = (float)zb[row * 64 + nG];
        float h  = (float)hb[row * 64 + nG];
        hb[row * 64 + nG] = (_Float16)((1.f - z) * h + z * hm);
      }
    }
    __syncthreads();
  }

  // ---- readout: rd = tanh(h @ W1 + b1), rows: R[0,16) O[16,176) D[176,352) ----
  for (int job = wid; job < 22; job += WAVES) {
    const float* b1;
    int rb;
    v8f c;
    if (job == 0) {
      rb = 0; b1 = p.b1R;
      c = gemm1<2>(hR, 64, 0, p.wt, T_W1_R, 1, 0, lane);
    } else if (job < 11) {
      int mt = job - 1; rb = 16 + mt * 16; b1 = p.b1O;
      c = gemm1<2>(hO, 64, mt * 16, p.wt, T_W1_O, 1, 0, lane);
    } else {
      int mt = job - 11; rb = 176 + mt * 16; b1 = p.b1D;
      c = gemm1<2>(hD, 64, mt * 16, p.wt, T_W1_D, 1, 0, lane);
    }
    int half = lane >> 4, n = lane & 15;
    float bv = b1[n];
#pragma unroll
    for (int r = 0; r < 8; r++) {
      int row = rb + r + 8 * half;
      rd[row * 16 + n] = fast_tanh(c[r] + bv);
    }
  }
  __syncthreads();

  // ---- per-batch scores ----
  if (tid < 16) {
    float acc = 0.f;
    for (int k = 0; k < 16; k++) acc += rd[tid * 16 + k] * p.w2R[k];
    p.sR[b0 + tid] = acc;
  } else if (tid < 32) {
    int bl = tid - 16; float acc = 0.f;
    for (int o = 0; o < 10; o++)
      for (int k = 0; k < 16; k++) acc += rd[(16 + bl * 10 + o) * 16 + k] * p.w2O[k];
    p.sO[b0 + bl] = acc;
  } else if (tid < 48) {
    int bl = tid - 32; float acc = 0.f;
    for (int d = 0; d < 11; d++)
      for (int k = 0; k < 16; k++) acc += rd[(176 + bl * 11 + d) * 16 + k] * p.w2D[k];
    p.sD[b0 + bl] = acc;
  }
}

// ---------------------------------------------------------------------------
// out[b, j] = sR[b]*tp0[j] + sO[b]*tp1[j] + sD[b]*tp2[j]
// ---------------------------------------------------------------------------
__global__ void ggnn_combine(const float* __restrict__ sR, const float* __restrict__ sO,
                             const float* __restrict__ sD, const float* __restrict__ tp,
                             float* __restrict__ out, int n) {
  int i = blockIdx.x * 256 + threadIdx.x;
  if (i >= n) return;
  int b = i / 60, j = i - b * 60;
  out[i] = sR[b] * tp[j] + sO[b] * tp[60 + j] + sD[b] * tp[120 + j];
}

extern "C" void kernel_launch(void* const* d_in, const int* in_sizes, int n_in,
                              void* d_out, int out_size, void* d_ws, size_t ws_size,
                              hipStream_t stream) {
  (void)in_sizes; (void)n_in; (void)ws_size;
  const float* X    = (const float*)d_in[0];
  const float* A    = (const float*)d_in[1];
  const float* WgR  = (const float*)d_in[2];  const float* bgR = (const float*)d_in[3];
  const float* WgO  = (const float*)d_in[4];  const float* bgO = (const float*)d_in[5];
  const float* WgD  = (const float*)d_in[6];  const float* bgD = (const float*)d_in[7];
  const float* WzR  = (const float*)d_in[8];  const float* WrR = (const float*)d_in[9];
  const float* WmR  = (const float*)d_in[10]; const float* bmR = (const float*)d_in[11];
  const float* WzO  = (const float*)d_in[12]; const float* WrO = (const float*)d_in[13];
  const float* WmO  = (const float*)d_in[14]; const float* bmO = (const float*)d_in[15];
  const float* WzD  = (const float*)d_in[16]; const float* WrD = (const float*)d_in[17];
  const float* WmD  = (const float*)d_in[18]; const float* bmD = (const float*)d_in[19];
  const float* W1R  = (const float*)d_in[20]; const float* b1R = (const float*)d_in[21];
  const float* W2R  = (const float*)d_in[22];
  const float* W1O  = (const float*)d_in[23]; const float* b1O = (const float*)d_in[24];
  const float* W2O  = (const float*)d_in[25];
  const float* W1D  = (const float*)d_in[26]; const float* b1D = (const float*)d_in[27];
  const float* W2D  = (const float*)d_in[28];
  const float* temporal = (const float*)d_in[29];
  const float* gk   = (const float*)d_in[30];
  const float* grk  = (const float*)d_in[31];
  const float* gb   = (const float*)d_in[32];
  const float* wo   = (const float*)d_in[33];
  const int*   idx  = (const int*)d_in[34];

  unsigned* wsw = (unsigned*)d_ws;
  float* sR = (float*)(wsw + T_TOTAL * 256);
  float* sO = sR + 8192;
  float* sD = sO + 8192;
  float* tp = sD + 8192;

  // temporal probs (independent, tiny)
  ggnn_temporal<<<1, 64, 0, stream>>>(temporal, gk, grk, gb, wo, idx, tp);

  // pack weights into WMMA B-tile layout (f16)
  struct PD { const float* s0; const float* s1; int K, KT, Ncols, NT, toff; };
  const PD pd[12] = {
    { WgR, nullptr,  40, 2, 64, 4, T_WG_R  },
    { WgO, nullptr,  70, 3, 64, 4, T_WG_O  },
    { WgD, nullptr,  70, 3, 64, 4, T_WG_D  },
    { WzR, WrR,     128, 4, 64, 8, T_WZR_R },
    { WzO, WrO,     192, 6, 64, 8, T_WZR_O },
    { WzD, WrD,     128, 4, 64, 8, T_WZR_D },
    { WmR, nullptr, 128, 4, 64, 4, T_WM_R  },
    { WmO, nullptr, 192, 6, 64, 4, T_WM_O  },
    { WmD, nullptr, 128, 4, 64, 4, T_WM_D  },
    { W1R, nullptr,  64, 2, 16, 1, T_W1_R  },
    { W1O, nullptr,  64, 2, 16, 1, T_W1_O  },
    { W1D, nullptr,  64, 2, 16, 1, T_W1_D  },
  };
  for (int i = 0; i < 12; i++) {
    int total = pd[i].KT * pd[i].NT * 256;
    int blocks = (total + 255) / 256;
    ggnn_pack_b<<<blocks, 256, 0, stream>>>(pd[i].s0, pd[i].s1, pd[i].K, pd[i].Ncols,
                                            pd[i].NT, wsw + pd[i].toff * 256, total);
  }

  // main GGNN
  MainParams mp;
  mp.X = X; mp.A = A;
  mp.bgR = bgR; mp.bgO = bgO; mp.bgD = bgD;
  mp.bmR = bmR; mp.bmO = bmO; mp.bmD = bmD;
  mp.b1R = b1R; mp.b1O = b1O; mp.b1D = b1D;
  mp.w2R = W2R; mp.w2O = W2O; mp.w2D = W2D;
  mp.wt = wsw;
  mp.sR = sR; mp.sO = sO; mp.sD = sD;
  ggnn_main<<<8192 / NB, 256, SMEM_BYTES, stream>>>(mp);

  // combine with temporal probs
  int n = out_size;  // 8192 * 60
  ggnn_combine<<<(n + 255) / 256, 256, 0, stream>>>(sR, sO, sD, tp, (float*)d_out, n);
}
